// ViT_1803886265636
// MI455X (gfx1250) — matrix operand order repaired and checked by reading, not compiled, not run on hardware
//
#include <hip/hip_runtime.h>
#include <hip/hip_bf16.h>
#include <math.h>
#include <stdint.h>

// ---------------------------------------------------------------------------
// ViT-Base/16 forward for gfx1250 (MI455X).
//  - All GEMMs via v_wmma_f32_16x16x32_f16 (f16 in, f32 accumulate)
//  - Global->LDS tiles via Tensor Data Mover (tensor_load_to_lds, TENSORcnt),
//    double-buffered so DMA of tile kt+1 overlaps WMMA on tile kt
//  - B matrices stored transposed (NxK) so A- and B-tiles are identical
//    row-major 2D tiles (TDM zero-fills OOB, applies LDS pitch padding)
// ---------------------------------------------------------------------------

typedef __attribute__((ext_vector_type(16))) _Float16 v16h;
typedef __attribute__((ext_vector_type(8)))  _Float16 v8h;
typedef __attribute__((ext_vector_type(8)))  float    v8f;
typedef __attribute__((ext_vector_type(4)))  unsigned int u32x4;
typedef __attribute__((ext_vector_type(8)))  int      i32x8;
typedef __attribute__((ext_vector_type(4)))  int      i32x4;

#define VB   32
#define VC   3
#define VIMG 224
#define VP   16
#define VD   768
#define VH   12
#define VHS  64
#define VL   12
#define VNT  197
#define VNP  196
#define VFF  3072
#define VNC  1000
#define SLD  208        // padded row stride for 197-wide rows

constexpr int BM = 128, BN = 128, BK = 64, KPAD = 8;
constexpr int BKP = BK + KPAD;   // 72-half LDS pitch (144 B, 16B-aligned)

#ifndef __has_builtin
#define __has_builtin(x) 0
#endif
#if defined(__gfx1250__) && __has_builtin(__builtin_amdgcn_tensor_load_to_lds)
#define VIT_USE_TDM 1
#else
#define VIT_USE_TDM 0
#endif

struct GemmParams {
  const _Float16* A;   long long ldA, sAo, sAi;   // A: M x K row-major
  const _Float16* Bt;  long long ldB, sBo, sBi;   // B transposed: N x K row-major
  float*          C;   long long ldC, sCo, sCi;
  _Float16*       C16; long long ldC16, sC16o, sC16i;
  const float*    bias;                           // per-column (N), may be null
  const float*    R;   long long ldR, sRo, sRi;   // fp32 residual, may be null
  int M, N, K, inner, flags;                      // flags bit0 = exact GELU
  float alpha;
};

#if VIT_USE_TDM
// One TDM 2D tile load: tileRows x 64 halfs (128B rows), row stride ld (elems),
// into LDS at ldsOff, padding 4 dwords after every 32 dwords (72-half pitch).
// Reads beyond tensor_dim return zero (free M/K-edge guarding).
__device__ __forceinline__ void tdm_load_tile(const _Float16* gptr, long long ld,
                                              int remK, int remRows, int tileRows,
                                              unsigned ldsOff) {
  const unsigned long long ga = (unsigned long long)(uintptr_t)gptr;
  u32x4 g0;
  g0[0] = 1u;                                          // count=1, user desc
  g0[1] = ldsOff;                                      // lds_addr
  g0[2] = (unsigned)(ga & 0xFFFFFFFFull);              // global_addr[31:0]
  g0[3] = (unsigned)((ga >> 32) & 0x01FFFFFFull)       // global_addr[56:32]
          | (2u << 30);                                // type = 2 (image)
  const unsigned td0 = (unsigned)(remK   > 0 ? remK   : 0);
  const unsigned td1 = (unsigned)(remRows > 0 ? remRows : 0);
  const unsigned long long s0 = (unsigned long long)ld;
  i32x8 g1;
  g1[0] = (int)((1u << 16)      // data_size = 2 bytes
              | (1u << 20)      // pad_enable
              | (4u << 22)      // pad_interval: every 32 dwords (128B row)
              | (3u << 25));    // pad_amount: 4 dwords (8 halfs)
  g1[1] = (int)((td0 & 0xFFFFu) << 16);                       // tensor_dim0 lo
  g1[2] = (int)((td0 >> 16) | ((td1 & 0xFFFFu) << 16));       // td0 hi | td1 lo
  g1[3] = (int)((td1 >> 16) | ((unsigned)BK << 16));          // td1 hi | tile_dim0
  g1[4] = (int)((unsigned)tileRows & 0xFFFFu);                // tile_dim1 (dim2=0)
  g1[5] = (int)(s0 & 0xFFFFFFFFull);                          // dim0_stride lo
  g1[6] = (int)((s0 >> 32) & 0xFFFFull);                      // dim0_stride hi
  g1[7] = 0;
  const i32x4 z4 = {0, 0, 0, 0};
#if __clang_major__ >= 23
  const i32x8 z8 = {0, 0, 0, 0, 0, 0, 0, 0};
  __builtin_amdgcn_tensor_load_to_lds(g0, g1, z4, z4, z8, 0);
#else
  __builtin_amdgcn_tensor_load_to_lds(g0, g1, z4, z4, 0);
#endif
}

__device__ __forceinline__ void tdm_wait0() {
#if __has_builtin(__builtin_amdgcn_s_wait_tensorcnt)
  __builtin_amdgcn_s_wait_tensorcnt(0);
#else
  asm volatile("s_wait_tensorcnt 0x0" ::: "memory");
#endif
  asm volatile("" ::: "memory");
}
#endif  // VIT_USE_TDM

// ---------------------------------------------------------------------------
// Generic f16 WMMA GEMM: C = epilogue(alpha * A(MxK) @ Bt(NxK)^T)
// Block tile 128x128, K-tile 64 (2 WMMA K-steps). 8 waves (4 M x 2 N),
// wave tile 32x64: 16 x v_wmma_f32_16x16x32_f16 per wave per K-tile.
// TDM double-buffered: DMA of tile kt+1 overlaps compute of tile kt.
// ---------------------------------------------------------------------------
__global__ __launch_bounds__(256) void vit_gemm_f16_wmma(GemmParams p) {
  __shared__ __align__(16) _Float16 As[2][BM][BKP];
  __shared__ __align__(16) _Float16 Bs[2][BN][BKP];

  const int bz = blockIdx.z;
  const int bo = bz / p.inner;
  const int bi = bz % p.inner;

  const _Float16* A  = p.A  + bo * p.sAo + bi * p.sAi;
  const _Float16* Bg = p.Bt + bo * p.sBo + bi * p.sBi;

  const int tid  = threadIdx.x;
  const int lane = tid & 31;
  const int wave = tid >> 5;
  const int wm   = wave & 3;     // 0..3 (M)
  const int wn   = wave >> 2;    // 0..1 (N)
  const int l16  = lane & 15;
  const int lhi  = lane >> 4;    // 0 or 1

  const int blockM = blockIdx.y * BM;
  const int blockN = blockIdx.x * BN;

  v8f acc[2][4] = {};

  const int kTiles = (p.K + BK - 1) / BK;

#if VIT_USE_TDM
  const unsigned aoff0 = (unsigned)(uintptr_t)(&As[0][0][0]);
  const unsigned aoff1 = (unsigned)(uintptr_t)(&As[1][0][0]);
  const unsigned boff0 = (unsigned)(uintptr_t)(&Bs[0][0][0]);
  const unsigned boff1 = (unsigned)(uintptr_t)(&Bs[1][0][0]);
  if (wave == 0) {   // prologue: tile 0 -> buffer 0
    tdm_load_tile(A,  p.ldA, p.K, p.M - blockM, BM, aoff0);
    tdm_load_tile(Bg, p.ldB, p.K, p.N - blockN, BN, boff0);
  }
#endif

  for (int kt = 0; kt < kTiles; ++kt) {
    const int k0 = kt * BK;

#if VIT_USE_TDM
    const int buf = kt & 1;
    if (wave == 0) tdm_wait0();          // tile kt DMA complete
    __syncthreads();                     // visible to all waves
    if (wave == 0 && (kt + 1) < kTiles) {
      const int k1 = k0 + BK;            // prefetch tile kt+1 -> other buffer
      tdm_load_tile(A  + (long long)blockM * p.ldA + k1, p.ldA,
                    p.K - k1, p.M - blockM, BM, buf ? aoff0 : aoff1);
      tdm_load_tile(Bg + (long long)blockN * p.ldB + k1, p.ldB,
                    p.K - k1, p.N - blockN, BN, buf ? boff0 : boff1);
      asm volatile("" ::: "memory");
    }
#else
    const int buf = 0;
    // --- software loader: 1 row per thread (128 A rows + 128 B rows) -------
    {
      const int r   = tid & 127;
      const bool doB = tid >= 128;
      const int gmn = (doB ? blockN : blockM) + r;
      const int lim = doB ? p.N : p.M;
      const _Float16* src =
          (doB ? Bg + (long long)gmn * p.ldB : A + (long long)gmn * p.ldA) + k0;
      _Float16* dst = doB ? &Bs[0][r][0] : &As[0][r][0];
      if (gmn < lim && (k0 + BK) <= p.K) {
#pragma unroll
        for (int i = 0; i < BK / 8; ++i)
          *(v8h*)(dst + i * 8) = *(const v8h*)(src + i * 8);
      } else {
#pragma unroll
        for (int i = 0; i < BK; ++i) {
          _Float16 v = (_Float16)0.0f;
          if (gmn < lim && (k0 + i) < p.K) v = src[i];
          dst[i] = v;
        }
      }
    }
    __syncthreads();
#endif

    // --- compute: 2 K-steps of 32 ------------------------------------------
    // A (16x32 f16): lanes 0-15 hold K {0..7,16..23}, lanes 16-31 {8..15,24..31}
    const int ga = lhi ? 8 : 0;
    // B (32x16 f16): lanes 0-15 hold K {0..7,16..23}, lanes 16-31 {8..15,24..31}
    const int gb = lhi ? 8 : 0;

#pragma unroll
    for (int ks = 0; ks < 2; ++ks) {
      v16h afrag[2], bfrag[4];
#pragma unroll
      for (int mi = 0; mi < 2; ++mi) {
        const int m = wm * 32 + mi * 16 + l16;
        const _Float16* ap = &As[buf][m][ks * 32];
        v8h lo = *(const v8h*)(ap + ga);
        v8h hi = *(const v8h*)(ap + ga + 16);
#pragma unroll
        for (int i = 0; i < 8; ++i) { afrag[mi][i] = lo[i]; afrag[mi][8 + i] = hi[i]; }
      }
#pragma unroll
      for (int ni = 0; ni < 4; ++ni) {
        const int n = wn * 64 + ni * 16 + l16;
        const _Float16* bp = &Bs[buf][n][ks * 32];
        v8h lo = *(const v8h*)(bp + gb);
        v8h hi = *(const v8h*)(bp + gb + 16);
#pragma unroll
        for (int i = 0; i < 8; ++i) { bfrag[ni][i] = lo[i]; bfrag[ni][8 + i] = hi[i]; }
      }
#pragma unroll
      for (int mi = 0; mi < 2; ++mi)
#pragma unroll
        for (int ni = 0; ni < 4; ++ni)
          acc[mi][ni] = __builtin_amdgcn_wmma_f32_16x16x32_f16(
              false, afrag[mi], false, bfrag[ni], (short)0, acc[mi][ni],
              false, false);
    }

#if !VIT_USE_TDM
    __syncthreads();
#endif
  }

  // --- epilogue --------------------------------------------------------------
  float*          Cp  = p.C   ? p.C   + bo * p.sCo   + bi * p.sCi   : nullptr;
  _Float16*       C16 = p.C16 ? p.C16 + bo * p.sC16o + bi * p.sC16i : nullptr;
  const float*    Rp  = p.R   ? p.R   + bo * p.sRo   + bi * p.sRi   : nullptr;

#pragma unroll
  for (int mi = 0; mi < 2; ++mi)
#pragma unroll
    for (int ni = 0; ni < 4; ++ni) {
      const int n = blockN + wn * 64 + ni * 16 + l16;
      if (n >= p.N) continue;
      const float bval = p.bias ? p.bias[n] : 0.0f;
#pragma unroll
      for (int r = 0; r < 8; ++r) {
        const int m = blockM + wm * 32 + mi * 16 + lhi * 8 + r;
        if (m >= p.M) continue;
        float v = acc[mi][ni][r] * p.alpha + bval;
        if (p.flags & 1) v = 0.5f * v * (1.0f + erff(v * 0.70710678118654752f));
        if (Rp)  v += Rp[(long long)m * p.ldR + n];
        if (Cp)  Cp[(long long)m * p.ldC + n] = v;
        if (C16) C16[(long long)m * p.ldC16 + n] = (_Float16)v;
      }
    }
}

// ---------------------------------------------------------------------------
// LayerNorm (ddof=1). One block per row. Optional fp32 out + f16 out.
// ---------------------------------------------------------------------------
__global__ __launch_bounds__(256) void vit_layernorm(
    const float* __restrict__ X, long long inRowStride,
    float* __restrict__ Y, _Float16* __restrict__ Y16,
    const float* __restrict__ g, const float* __restrict__ b, int D) {
  __shared__ float red[256];
  const int tid = threadIdx.x;
  const float* x = X + (long long)blockIdx.x * inRowStride;

  float s = 0.0f;
  for (int i = tid; i < D; i += 256) s += x[i];
  red[tid] = s; __syncthreads();
  for (int st = 128; st > 0; st >>= 1) {
    if (tid < st) red[tid] += red[tid + st];
    __syncthreads();
  }
  const float mean = red[0] / (float)D;
  __syncthreads();

  float s2 = 0.0f;
  for (int i = tid; i < D; i += 256) { float d = x[i] - mean; s2 += d * d; }
  red[tid] = s2; __syncthreads();
  for (int st = 128; st > 0; st >>= 1) {
    if (tid < st) red[tid] += red[tid + st];
    __syncthreads();
  }
  const float var = red[0] / (float)(D - 1);
  const float rstd = rsqrtf(var + 1e-5f);

  for (int i = tid; i < D; i += 256) {
    const float y = g[i] * (x[i] - mean) * rstd + b[i];
    if (Y)   Y[(long long)blockIdx.x * D + i] = y;
    if (Y16) Y16[(long long)blockIdx.x * D + i] = (_Float16)y;
  }
}

// ---------------------------------------------------------------------------
// Softmax over a 197-wide row (stride SLD); writes normalized f16, zero pad.
// ---------------------------------------------------------------------------
__global__ __launch_bounds__(256) void vit_softmax(
    const float* __restrict__ S, _Float16* __restrict__ P, int n, int ld) {
  __shared__ float red[256];
  const int tid = threadIdx.x;
  const float* row = S + (long long)blockIdx.x * ld;
  _Float16* prow = P + (long long)blockIdx.x * ld;

  float m = -3.0e38f;
  for (int i = tid; i < n; i += 256) m = fmaxf(m, row[i]);
  red[tid] = m; __syncthreads();
  for (int st = 128; st > 0; st >>= 1) {
    if (tid < st) red[tid] = fmaxf(red[tid], red[tid + st]);
    __syncthreads();
  }
  const float mx = red[0];
  __syncthreads();

  float s = 0.0f;
  for (int i = tid; i < n; i += 256) s += __expf(row[i] - mx);
  red[tid] = s; __syncthreads();
  for (int st = 128; st > 0; st >>= 1) {
    if (tid < st) red[tid] += red[tid + st];
    __syncthreads();
  }
  const float inv = 1.0f / red[0];

  for (int i = tid; i < ld; i += 256) {
    float v = (i < n) ? __expf(row[i] - mx) * inv : 0.0f;
    prow[i] = (_Float16)v;
  }
}

// ---------------------------------------------------------------------------
// Patch gather: x(B,C,224,224) -> xp16(B*196, 768) f16
// ---------------------------------------------------------------------------
__global__ __launch_bounds__(256) void vit_gather_patches(
    const float* __restrict__ x, _Float16* __restrict__ xp) {
  const long long idx = (long long)blockIdx.x * 256 + threadIdx.x;
  const long long total = (long long)VB * VNP * VD;
  if (idx >= total) return;
  const int col = (int)(idx % VD);
  const long long r = idx / VD;
  const int b = (int)(r / VNP);
  const int pidx = (int)(r % VNP);
  const int gy = pidx / 14, gx = pidx % 14;
  const int c = col / (VP * VP);
  const int rem = col % (VP * VP);
  const int py = rem / VP, px = rem % VP;
  const long long src =
      (((long long)b * VC + c) * VIMG + (gy * VP + py)) * VIMG + (gx * VP + px);
  xp[idx] = (_Float16)x[src];
}

// cls row: t[b,0,:] = cls_token + pos_emb[0]
__global__ __launch_bounds__(256) void vit_cls_pos(
    const float* __restrict__ cls, const float* __restrict__ pos,
    float* __restrict__ t) {
  const int idx = blockIdx.x * 256 + threadIdx.x;
  if (idx >= VB * VD) return;
  const int b = idx / VD, d = idx % VD;
  t[(long long)b * VNT * VD + d] = cls[d] + pos[d];
}

// fp32 (L,K,N) -> f16 transposed (L,N,K)
__global__ __launch_bounds__(256) void vit_cvt_t_f16(
    const float* __restrict__ in, _Float16* __restrict__ outp,
    int K, int N, long long total) {
  const long long idx = (long long)blockIdx.x * 256 + threadIdx.x;
  if (idx >= total) return;
  const long long KN = (long long)K * N;
  const int l = (int)(idx / KN);
  const long long rem = idx % KN;
  const int n = (int)(rem / K);
  const int k = (int)(rem % K);
  outp[idx] = (_Float16)in[(long long)l * KN + (long long)k * N + n];
}

// Wq/Wk/Wv (L,H,D,HS) -> f16 transposed (L, 2304, 768): [l][col][d]
__global__ __launch_bounds__(256) void vit_build_wqkv_t(
    const float* __restrict__ Wq, const float* __restrict__ Wk,
    const float* __restrict__ Wv, _Float16* __restrict__ outp) {
  const long long idx = (long long)blockIdx.x * 256 + threadIdx.x;
  const long long total = (long long)VL * 3 * VD * VD;
  if (idx >= total) return;
  const int d = (int)(idx % VD);
  const long long r = idx / VD;
  const int col = (int)(r % (3 * VD));
  const int l = (int)(r / (3 * VD));
  const float* src; int cc;
  if (col < VD)          { src = Wq; cc = col; }
  else if (col < 2 * VD) { src = Wk; cc = col - VD; }
  else                   { src = Wv; cc = col - 2 * VD; }
  const int h = cc / VHS, e = cc % VHS;
  outp[idx] = (_Float16)src[((((long long)l * VH + h) * VD) + d) * VHS + e];
}

// concat bq|bk|bv per layer -> (L, 2304) fp32
__global__ __launch_bounds__(256) void vit_build_bqkv(
    const float* __restrict__ bq, const float* __restrict__ bk,
    const float* __restrict__ bv, float* __restrict__ outp) {
  const int idx = blockIdx.x * 256 + threadIdx.x;
  if (idx >= VL * 3 * VD) return;
  const int c = idx % (3 * VD);
  const int l = idx / (3 * VD);
  float v;
  if (c < VD)          v = bq[(long long)l * VD + c];
  else if (c < 2 * VD) v = bk[(long long)l * VD + (c - VD)];
  else                 v = bv[(long long)l * VD + (c - 2 * VD)];
  outp[idx] = v;
}

// qkv fp32 (B*197, 2304) -> q16 (B,H,197,64), k16 (B,H,197,64), vT16 (B,H,64,SLD)
__global__ __launch_bounds__(256) void vit_reshape_qkv(
    const float* __restrict__ qkv, _Float16* __restrict__ q16,
    _Float16* __restrict__ k16, _Float16* __restrict__ vT16) {
  const long long idx = (long long)blockIdx.x * 256 + threadIdx.x;
  const long long total = (long long)VB * VNT * VD;
  if (idx >= total) return;
  const int col = (int)(idx % VD);
  const long long row = idx / VD;
  const int b = (int)(row / VNT);
  const int n = (int)(row % VNT);
  const int h = col / VHS, e = col % VHS;
  const long long base = row * (3LL * VD);
  const long long bh = (long long)b * VH + h;
  q16[(bh * VNT + n) * VHS + e]  = (_Float16)qkv[base + col];
  k16[(bh * VNT + n) * VHS + e]  = (_Float16)qkv[base + VD + col];
  vT16[(bh * VHS + e) * SLD + n] = (_Float16)qkv[base + 2 * VD + col];
}

// ---------------------------------------------------------------------------
// Host side
// ---------------------------------------------------------------------------
static inline GemmParams gemm_init() {
  GemmParams p{};
  p.inner = 1;
  p.alpha = 1.0f;
  return p;
}

extern "C" void kernel_launch(void* const* d_in, const int* in_sizes, int n_in,
                              void* d_out, int out_size, void* d_ws, size_t ws_size,
                              hipStream_t stream) {
  const float* x    = (const float*)d_in[0];
  const float* Wp   = (const float*)d_in[1];
  const float* bp   = (const float*)d_in[2];
  const float* pos  = (const float*)d_in[3];
  const float* cls  = (const float*)d_in[4];
  const float* Wq   = (const float*)d_in[5];
  const float* bq   = (const float*)d_in[6];
  const float* Wk   = (const float*)d_in[7];
  const float* bk   = (const float*)d_in[8];
  const float* Wv   = (const float*)d_in[9];
  const float* bv   = (const float*)d_in[10];
  const float* Wo   = (const float*)d_in[11];
  const float* bo   = (const float*)d_in[12];
  const float* g1   = (const float*)d_in[13];
  const float* b1   = (const float*)d_in[14];
  const float* g2   = (const float*)d_in[15];
  const float* b2   = (const float*)d_in[16];
  const float* W1   = (const float*)d_in[17];
  const float* bf1  = (const float*)d_in[18];
  const float* W2   = (const float*)d_in[19];
  const float* bf2  = (const float*)d_in[20];
  const float* gf   = (const float*)d_in[21];
  const float* betaf= (const float*)d_in[22];
  const float* Wh   = (const float*)d_in[23];
  const float* bh   = (const float*)d_in[24];
  float* out = (float*)d_out;
  (void)in_sizes; (void)n_in; (void)out_size; (void)ws_size;

  char* wsp = (char*)d_ws;
  auto alloc = [&](size_t bytes) -> void* {
    void* p = (void*)wsp;
    wsp += (bytes + 255) & ~(size_t)255;
    return p;
  };

  const long long M  = (long long)VB * VNT;   // 6304
  const long long BH = (long long)VB * VH;    // 384

  // transposed (NxK) f16 weights
  _Float16* wp16T   = (_Float16*)alloc((size_t)VD * VD * 2);
  _Float16* wqkv16T = (_Float16*)alloc((size_t)VL * 3 * VD * VD * 2);
  _Float16* wo16T   = (_Float16*)alloc((size_t)VL * VD * VD * 2);
  _Float16* w116T   = (_Float16*)alloc((size_t)VL * VFF * VD * 2);
  _Float16* w216T   = (_Float16*)alloc((size_t)VL * VD * VFF * 2);
  _Float16* wh16T   = (_Float16*)alloc((size_t)VNC * VD * 2);
  float*    bqkvf   = (float*)   alloc((size_t)VL * 3 * VD * 4);

  _Float16* xp16   = (_Float16*)alloc((size_t)VB * VNP * VD * 2);
  float*    t      = (float*)   alloc((size_t)M * VD * 4);
  float*    h      = (float*)   alloc((size_t)M * VD * 4);
  _Float16* h16    = (_Float16*)alloc((size_t)M * VD * 2);
  float*    qkv    = (float*)   alloc((size_t)M * 3 * VD * 4);
  _Float16* q16    = (_Float16*)alloc((size_t)BH * VNT * VHS * 2);
  _Float16* k16    = (_Float16*)alloc((size_t)BH * VNT * VHS * 2);
  _Float16* vT16   = (_Float16*)alloc((size_t)BH * VHS * SLD * 2);
  float*    scores = (float*)   alloc((size_t)BH * VNT * SLD * 4);
  _Float16* att16  = (_Float16*)alloc((size_t)BH * VNT * SLD * 2);
  _Float16* acat16 = (_Float16*)alloc((size_t)M * VD * 2);
  float*    hres   = (float*)   alloc((size_t)M * VD * 4);
  float*    h2     = (float*)   alloc((size_t)M * VD * 4);
  _Float16* h2_16  = (_Float16*)alloc((size_t)M * VD * 2);
  _Float16* mid16  = (_Float16*)alloc((size_t)M * VFF * 2);
  _Float16* cls16  = (_Float16*)alloc((size_t)VB * VD * 2);

  auto blocks = [](long long n) { return (unsigned)((n + 255) / 256); };
  auto gemm = [&](const GemmParams& p, int batches) {
    dim3 g((p.N + BN - 1) / BN, (p.M + BM - 1) / BM, batches);
    vit_gemm_f16_wmma<<<g, 256, 0, stream>>>(p);
  };

  // ---- weight conversion (to transposed NxK f16) ---------------------------
  vit_cvt_t_f16<<<blocks((long long)VD * VD), 256, 0, stream>>>(
      Wp, wp16T, VD, VD, (long long)VD * VD);
  vit_build_wqkv_t<<<blocks((long long)VL * 3 * VD * VD), 256, 0, stream>>>(
      Wq, Wk, Wv, wqkv16T);
  vit_build_bqkv<<<blocks((long long)VL * 3 * VD), 256, 0, stream>>>(bq, bk, bv, bqkvf);
  vit_cvt_t_f16<<<blocks((long long)VL * VD * VD), 256, 0, stream>>>(
      Wo, wo16T, VD, VD, (long long)VL * VD * VD);
  vit_cvt_t_f16<<<blocks((long long)VL * VD * VFF), 256, 0, stream>>>(
      W1, w116T, VD, VFF, (long long)VL * VD * VFF);
  vit_cvt_t_f16<<<blocks((long long)VL * VFF * VD), 256, 0, stream>>>(
      W2, w216T, VFF, VD, (long long)VL * VFF * VD);
  vit_cvt_t_f16<<<blocks((long long)VD * VNC), 256, 0, stream>>>(
      Wh, wh16T, VD, VNC, (long long)VD * VNC);

  // ---- patch embed: t[b,1..196,:] = patches @ Wp + bp + pos[1..196] --------
  vit_gather_patches<<<blocks((long long)VB * VNP * VD), 256, 0, stream>>>(x, xp16);
  vit_cls_pos<<<blocks((long long)VB * VD), 256, 0, stream>>>(cls, pos, t);
  {
    GemmParams p = gemm_init();
    p.A = xp16;   p.ldA = VD;  p.sAo = (long long)VNP * VD;
    p.Bt = wp16T; p.ldB = VD;
    p.C = t + VD; p.ldC = VD;  p.sCo = (long long)VNT * VD;
    p.bias = bp;
    p.R = pos + VD; p.ldR = VD;   // broadcast over batch (stride 0)
    p.M = VNP; p.N = VD; p.K = VD;
    gemm(p, VB);
  }

  // ---- transformer layers --------------------------------------------------
  for (int l = 0; l < VL; ++l) {
    vit_layernorm<<<(unsigned)M, 256, 0, stream>>>(t, VD, h, h16,
                                                   g1 + (long long)l * VD,
                                                   b1 + (long long)l * VD, VD);
    {   // qkv = h16 @ Wqkv[l] + bqkv[l]
      GemmParams p = gemm_init();
      p.A = h16; p.ldA = VD;
      p.Bt = wqkv16T + (long long)l * 3 * VD * VD; p.ldB = VD;
      p.C = qkv; p.ldC = 3 * VD;
      p.bias = bqkvf + (long long)l * 3 * VD;
      p.M = (int)M; p.N = 3 * VD; p.K = VD;
      gemm(p, 1);
    }
    vit_reshape_qkv<<<blocks(M * VD), 256, 0, stream>>>(qkv, q16, k16, vT16);

    {   // scores = 0.125 * q @ k^T  (Bt = k natural layout)
      GemmParams p = gemm_init();
      p.A = q16;  p.ldA = VHS; p.sAo = (long long)VNT * VHS;
      p.Bt = k16; p.ldB = VHS; p.sBo = (long long)VNT * VHS;
      p.C = scores; p.ldC = SLD; p.sCo = (long long)VNT * SLD;
      p.alpha = 0.125f;
      p.M = VNT; p.N = VNT; p.K = VHS;
      gemm(p, (int)BH);
    }
    vit_softmax<<<(unsigned)(BH * VNT), 256, 0, stream>>>(scores, att16, VNT, SLD);

    {   // acat[b, n, h*64+e] = att @ v  (Bt = v transposed: (64, 197-pad))
      GemmParams p = gemm_init();
      p.A = att16; p.ldA = SLD;
      p.sAo = (long long)VH * VNT * SLD; p.sAi = (long long)VNT * SLD;
      p.Bt = vT16; p.ldB = SLD;
      p.sBo = (long long)VH * VHS * SLD; p.sBi = (long long)VHS * SLD;
      p.C16 = acat16; p.ldC16 = VD;
      p.sC16o = (long long)VNT * VD; p.sC16i = VHS;
      p.inner = VH;
      p.M = VNT; p.N = VHS; p.K = VNT;
      gemm(p, (int)BH);
    }
    {   // hres = h + acat @ Wo[l] + bo[l]
      GemmParams p = gemm_init();
      p.A = acat16; p.ldA = VD;
      p.Bt = wo16T + (long long)l * VD * VD; p.ldB = VD;
      p.C = hres; p.ldC = VD;
      p.bias = bo + (long long)l * VD;
      p.R = h; p.ldR = VD;
      p.M = (int)M; p.N = VD; p.K = VD;
      gemm(p, 1);
    }
    vit_layernorm<<<(unsigned)M, 256, 0, stream>>>(hres, VD, h2, h2_16,
                                                   g2 + (long long)l * VD,
                                                   b2 + (long long)l * VD, VD);
    {   // mid16 = gelu(h2 @ W1[l] + bf1[l])
      GemmParams p = gemm_init();
      p.A = h2_16; p.ldA = VD;
      p.Bt = w116T + (long long)l * VFF * VD; p.ldB = VD;
      p.C16 = mid16; p.ldC16 = VFF;
      p.bias = bf1 + (long long)l * VFF;
      p.flags = 1;
      p.M = (int)M; p.N = VFF; p.K = VD;
      gemm(p, 1);
    }
    {   // t = h2 + mid @ W2[l] + bf2[l]
      GemmParams p = gemm_init();
      p.A = mid16; p.ldA = VFF;
      p.Bt = w216T + (long long)l * VD * VFF; p.ldB = VFF;
      p.C = t; p.ldC = VD;
      p.bias = bf2 + (long long)l * VD;
      p.R = h2; p.ldR = VD;
      p.M = (int)M; p.N = VD; p.K = VFF;
      gemm(p, 1);
    }
  }

  // ---- head: LN(cls rows) @ Wh + bh ----------------------------------------
  vit_layernorm<<<VB, 256, 0, stream>>>(t, (long long)VNT * VD, nullptr, cls16,
                                        gf, betaf, VD);
  {
    GemmParams p = gemm_init();
    p.A = cls16;  p.ldA = VD;
    p.Bt = wh16T; p.ldB = VD;
    p.C = out; p.ldC = VNC;
    p.bias = bh;
    p.M = VB; p.N = VNC; p.K = VD;
    gemm(p, 1);
  }
}
